// HeteroLinear_5703716569224
// MI455X (gfx1250) — matrix operand order repaired and checked by reading, not compiled, not run on hardware
//
#include <hip/hip_runtime.h>
#include <hip/hip_bf16.h>

typedef __attribute__((ext_vector_type(16))) __bf16 v16bf;
typedef __attribute__((ext_vector_type(8)))  float  v8f;
typedef __attribute__((ext_vector_type(4)))  float  v4f;
typedef int hl_v4i __attribute__((vector_size(16)));

#define IN_DIM  1024
#define OUT_DIM 1024
#define NTYPES  8
#define NTOK    16384

#define AS1 __attribute__((address_space(1)))
#define AS3 __attribute__((address_space(3)))

#if defined(__has_builtin)
#if __has_builtin(__builtin_amdgcn_global_load_async_to_lds_b128)
#define HL_ASYNC_STAGE 1
#endif
#endif

__device__ __forceinline__ void hl_wait_async0() {
#if defined(__has_builtin) && __has_builtin(__builtin_amdgcn_s_wait_asynccnt)
    __builtin_amdgcn_s_wait_asynccnt(0);
#else
    asm volatile("s_wait_asynccnt 0x0" ::: "memory");
#endif
}

// async global->LDS copy of one 32B fragment (two 16B beats); offset applies to both addresses
__device__ __forceinline__ void hl_stage32(const __bf16* gsrc, v16bf* ldst) {
#ifdef HL_ASYNC_STAGE
    __bf16* gnc = const_cast<__bf16*>(gsrc);
    hl_v4i AS1* gp = (hl_v4i AS1*)gnc;
    hl_v4i AS3* lp = (hl_v4i AS3*)ldst;
    __builtin_amdgcn_global_load_async_to_lds_b128(gp, lp, 0,  0);
    __builtin_amdgcn_global_load_async_to_lds_b128(gp, lp, 16, 0);
#else
    *ldst = *(const v16bf*)gsrc;
#endif
}

// ---------------------------------------------------------------------------
// Repack W [T, IN, OUT] fp32 -> Wb bf16 in WMMA B-fragment order:
//   Wb[t][jG (OUT/16=64)][kc (IN/32=32)][lane (32)][16 halfs]
// B 32x16 bf16 order used here (contiguous k, A matches): lane n holds col n%16; lanes 0-15 carry
// K = kc*32 + 0..15, lanes 16-31 carry K = kc*32 + 16..31 (2 halfs / VGPR).
// ---------------------------------------------------------------------------
__global__ __launch_bounds__(256) void hl_repack_w(const float* __restrict__ W,
                                                   __bf16* __restrict__ Wb) {
    int tid = blockIdx.x * blockDim.x + threadIdx.x;   // 0 .. 524287
    int n   =  tid        & 31;
    int kc  = (tid >> 5)  & 31;
    int j   = (tid >> 10) & 63;
    int t   =  tid >> 16;
    int col   = j * 16 + (n & 15);
    int kbase = kc * 32 + ((n >> 4) << 4);
    const float* src = W + ((size_t)t * IN_DIM + kbase) * OUT_DIM + col;
    __bf16* dst = Wb + (size_t)tid * 16;               // 32B contiguous / thread
    #pragma unroll
    for (int p = 0; p < 16; ++p)
        dst[p] = (__bf16)src[(size_t)p * OUT_DIM];
}

__global__ void hl_zero_counts(int* counts) {
    if (threadIdx.x < NTYPES) counts[threadIdx.x] = 0;
}

__global__ __launch_bounds__(256) void hl_bucket(const int* __restrict__ types,
                                                 int* counts,
                                                 int* __restrict__ indices) {
    int i = blockIdx.x * blockDim.x + threadIdx.x;
    if (i >= NTOK) return;
    int t = types[i];
    int slot = atomicAdd(&counts[t], 1);
    indices[t * NTOK + slot] = i;
}

// ---------------------------------------------------------------------------
// Grouped GEMM: block = 8 waves, tile = 128 tokens x 128 cols for one type.
// Wave w owns row subtile w (16 gathered tokens) and all 8 col subtiles.
// B fragments staged via async global->LDS (ASYNCcnt), double-buffered.
// ---------------------------------------------------------------------------
__global__ __launch_bounds__(256) void hl_gemm(const float*  __restrict__ x,
                                               const float*  __restrict__ bias,
                                               const __bf16* __restrict__ Wb,
                                               const int*    __restrict__ counts,
                                               const int*    __restrict__ indices,
                                               float*        __restrict__ out) {
    const int t       = blockIdx.z;
    const int cnt     = counts[t];
    const int tileRow = blockIdx.y;                    // 128-token tiles
    if (tileRow * 128 >= cnt) return;                  // uniform early-exit
    const int colBlk  = blockIdx.x;                    // 0..7 -> 128 cols each
    const int w       = threadIdx.x >> 5;              // wave id 0..7
    const int lane    = threadIdx.x & 31;
    const int r16     = lane & 15;
    const int halfOfs = (lane >> 4) << 3;              // 0 or 8

    // ---- B staging: double-buffered fragment store, 16 KB LDS -------------
    __shared__ v16bf ldsB[2][8][32];                   // [buf][j][lane]
    const size_t kcStrideE = (size_t)32 * 16;          // elems per k-chunk
    const __bf16* wsrc = Wb + ((((size_t)t * 64 + colBlk * 8 + w) * 32) * 32
                               + lane) * 16;           // + kc*kcStrideE

    // ---- A gather: wave w's 16 tokens -------------------------------------
    const int rowT = tileRow * 128 + w * 16;
    int g  = rowT + r16;
    int gi = g < cnt ? g : cnt - 1;                    // clamp (cnt>0 here)
    const int token = indices[t * NTOK + gi];
    const float* xrow = x + (size_t)token * IN_DIM + 2 * halfOfs;

    v8f acc[8] = {};

    // prologue: stage kc = 0
    hl_stage32(wsrc, &ldsB[0][w][lane]);
#ifdef HL_ASYNC_STAGE
    hl_wait_async0();
#endif
    __syncthreads();

    for (int kc = 0; kc < 32; ++kc) {
        const int cur = kc & 1;

        // stage next K-chunk into the other buffer (overlaps with WMMA)
        if (kc < 31)
            hl_stage32(wsrc + (size_t)(kc + 1) * kcStrideE,
                       &ldsB[cur ^ 1][w][lane]);

        // A fragment: 16 contiguous floats, packed-convert to bf16
        const float* xp = xrow + kc * 32;
        __builtin_prefetch(xp + 64, 0, 0);             // global_prefetch_b8
        v4f f0 = *(const v4f*)(xp);
        v4f f1 = *(const v4f*)(xp + 4);
        v4f g0 = *(const v4f*)(xp + 8);
        v4f g1 = *(const v4f*)(xp + 12);
        v16bf a;
        #pragma unroll
        for (int e = 0; e < 4; ++e) {                  // elems 0-7 : K=kb+0..7
            a[e]      = (__bf16)f0[e];                 // elems 8-15: K=kb+8..15
            a[e + 4]  = (__bf16)f1[e];
            a[e + 8]  = (__bf16)g0[e];
            a[e + 12] = (__bf16)g1[e];
        }

        // 8 WMMAs from LDS fragments, staged in groups of 4
        v16bf bf[4];
        #pragma unroll
        for (int jj = 0; jj < 4; ++jj) bf[jj] = ldsB[cur][jj][lane];
        #pragma unroll
        for (int jj = 0; jj < 4; ++jj)
            acc[jj] = __builtin_amdgcn_wmma_f32_16x16x32_bf16(
                          false, a, false, bf[jj], (short)0, acc[jj], false, false);
        #pragma unroll
        for (int jj = 0; jj < 4; ++jj) bf[jj] = ldsB[cur][4 + jj][lane];
        #pragma unroll
        for (int jj = 0; jj < 4; ++jj)
            acc[4 + jj] = __builtin_amdgcn_wmma_f32_16x16x32_bf16(
                          false, a, false, bf[jj], (short)0, acc[4 + jj], false, false);

#ifdef HL_ASYNC_STAGE
        hl_wait_async0();                              // staged data in LDS
#endif
        __syncthreads();                               // ping-pong safe point
    }

    // Epilogue: C layout (ISA 7.12.2): lane L holds col L%16; VGPR r holds
    // row r (lanes 0-15) or r+8 (lanes 16-31). Scatter rows by token id.
    int  tokR[8];
    bool valR[8];
    #pragma unroll
    for (int r = 0; r < 8; ++r) {
        int gg  = rowT + r + halfOfs;
        valR[r] = gg < cnt;
        tokR[r] = indices[t * NTOK + (valR[r] ? gg : 0)];
    }
    #pragma unroll
    for (int j = 0; j < 8; ++j) {
        int col  = colBlk * 128 + j * 16 + r16;
        float bv = bias[t * OUT_DIM + col];
        #pragma unroll
        for (int r = 0; r < 8; ++r) {
            if (valR[r])
                out[(size_t)tokR[r] * OUT_DIM + col] = acc[j][r] + bv;
        }
    }
}

extern "C" void kernel_launch(void* const* d_in, const int* in_sizes, int n_in,
                              void* d_out, int out_size, void* d_ws, size_t ws_size,
                              hipStream_t stream) {
    const float* x     = (const float*)d_in[0];  // [4,4096,1024]
    const int*   types = (const int*)  d_in[1];  // [4,4096]
    const float* W     = (const float*)d_in[2];  // [8,1024,1024]
    const float* bias  = (const float*)d_in[3];  // [8,1024]
    float* out = (float*)d_out;                  // [4,4096,1024]

    char* ws       = (char*)d_ws;
    int*  counts   = (int*)ws;                                   // 32 B (+pad)
    int*  indices  = (int*)(ws + 64);                            // 512 KB
    __bf16* Wb     = (__bf16*)(ws + 64 + NTYPES * NTOK * sizeof(int)); // 16 MB

    // 1) repack W -> bf16 fragment layout (524288 threads, 16 halfs each)
    hl_repack_w<<<2048, 256, 0, stream>>>(W, Wb);
    // 2) bucket tokens by type
    hl_zero_counts<<<1, 32, 0, stream>>>(counts);
    hl_bucket<<<NTOK / 256, 256, 0, stream>>>(types, counts, indices);
    // 3) grouped GEMM: grid = (col blocks, 128-token tiles, types)
    dim3 grid(OUT_DIM / 128, NTOK / 128, NTYPES);
    hl_gemm<<<grid, 256, 0, stream>>>(x, bias, Wb, counts, indices, out);
}
